// Attention_34548716929382
// MI455X (gfx1250) — matrix operand order repaired and checked by reading, not compiled, not run on hardware
//
#include <hip/hip_runtime.h>
#include <hip/hip_bf16.h>
#include <math.h>

// ---------------------------------------------------------------------------
// Types for CDNA5 WMMA bf16 path
// ---------------------------------------------------------------------------
typedef __bf16 bf16_t;
typedef __attribute__((ext_vector_type(16))) __bf16 v16bf;
typedef __attribute__((ext_vector_type(8)))  __bf16 v8bf;
typedef __attribute__((ext_vector_type(8)))  float   v8f;

union V16U { v16bf v; v8bf h[2]; };

#define SEQ_LEN 4096
#define DMODEL  1024

// ---------------------------------------------------------------------------
// Elementwise f32 -> bf16 convert
// ---------------------------------------------------------------------------
__global__ void cvt_bf16_kernel(const float* __restrict__ in,
                                bf16_t* __restrict__ out, int n) {
    int i = blockIdx.x * blockDim.x + threadIdx.x;
    if (i < n) out[i] = (bf16_t)in[i];
}

// W[R,C] f32 row-major -> WT[C,R] bf16 row-major (so B operand is K-contiguous)
__global__ void transpose_cvt_kernel(const float* __restrict__ W,
                                     bf16_t* __restrict__ WT, int R, int C) {
    int i = blockIdx.x * blockDim.x + threadIdx.x;
    if (i >= R * C) return;
    int r = i / C, c = i - r * C;
    WT[(size_t)c * R + r] = (bf16_t)W[(size_t)r * C + c];
}

// ---------------------------------------------------------------------------
// Register-blocked bf16 WMMA GEMM:  C[M,N] (f32 accum) = A[M,K] @ BT[N,K]^T
// Each wave computes a 64x32 macro-tile = 4x2 tiles of 16x16 via
// v_wmma_f32_16x16x32_bf16. Per K-step: 12 b128 loads (6 fragments) feed 8 WMMAs
// (~21 FLOP/B from cache), 8 independent accumulators avoid WMMA RAW stalls,
// and register use (~150 VGPRs) stays well inside the 256 budget (no spills).
// K-loop is kept at unroll(1): one iteration already carries enough ILP and
// larger unrolls multiply live fragment registers into spill territory.
// Epilogue: *scale, +bias[col], +resid[row,col], store f32 and/or bf16
// (optionally transposed bf16 store: outB[col*M + row]).
// ---------------------------------------------------------------------------
__global__ void __launch_bounds__(256, 1) gemm_wmma_bf16_kernel(
    const bf16_t* __restrict__ A,  int lda,
    const bf16_t* __restrict__ BT, int ldb,
    const float*  __restrict__ bias,
    const float*  __restrict__ resid,
    float*        __restrict__ outF,
    bf16_t*       __restrict__ outB, int transB,
    int M, int N, int K, float scale)
{
    const int wave   = blockIdx.x * (blockDim.x >> 5) + (threadIdx.x >> 5);
    const int wavesN = N >> 5;                 // 32-wide macro-tiles in N
    const int twm = wave / wavesN;
    const int twn = wave - twm * wavesN;
    if (twm >= (M >> 6)) return;               // wave-uniform: EXEC stays all-ones

    const int lane = threadIdx.x & 31;
    const int lo   = lane & 15;                // M-row (A) / N-col (B) within tile
    const int hi   = lane >> 4;                // K half selector

    const int row0 = twm * 64;
    const int col0 = twn * 32;

    const bf16_t* arow[4];
    const bf16_t* brow[2];
#pragma unroll
    for (int i = 0; i < 4; ++i)
        arow[i] = A + (size_t)(row0 + i * 16 + lo) * lda;
#pragma unroll
    for (int j = 0; j < 2; ++j)
        brow[j] = BT + (size_t)(col0 + j * 16 + lo) * ldb;

    v8f acc[4][2];
#pragma unroll
    for (int i = 0; i < 4; ++i)
#pragma unroll
        for (int j = 0; j < 2; ++j) acc[i][j] = (v8f){};

#pragma unroll 1
    for (int k0 = 0; k0 < K; k0 += 32) {
        V16U a[4];
        v16bf b[2];
#pragma unroll
        for (int i = 0; i < 4; ++i) {
            // A 16x32 bf16 fragment: lanes 0-15 K=[0..7,16..23], lanes 16-31 K=[8..15,24..31]
            a[i].h[0] = *(const v8bf*)(arow[i] + k0 + hi * 8);
            a[i].h[1] = *(const v8bf*)(arow[i] + k0 + 16 + hi * 8);
        }
#pragma unroll
        for (int j = 0; j < 2; ++j) {
            // B 32x16 bf16 fragment: lanes 0-15 K=[0..7,16..23], lanes 16-31 K=[8..15,24..31]
            v8bf b0 = *(const v8bf*)(brow[j] + k0 + hi * 8);
            v8bf b1 = *(const v8bf*)(brow[j] + k0 + 16 + hi * 8);
            b[j] = __builtin_shufflevector(b0, b1, 0,1,2,3,4,5,6,7,8,9,10,11,12,13,14,15);
        }
#pragma unroll
        for (int i = 0; i < 4; ++i)
#pragma unroll
            for (int j = 0; j < 2; ++j)
                acc[i][j] = __builtin_amdgcn_wmma_f32_16x16x32_bf16(
                                false, a[i].v, false, b[j], (short)0,
                                acc[i][j], false, false);
    }

#pragma unroll
    for (int j = 0; j < 2; ++j) {
        const int col  = col0 + j * 16 + lo;
        const float ba = bias ? bias[col] : 0.0f;
#pragma unroll
        for (int i = 0; i < 4; ++i) {
#pragma unroll
            for (int r = 0; r < 8; ++r) {
                const int row = row0 + i * 16 + r + hi * 8;  // C/D: vgpr r -> M = r + 8*hi
                float v = acc[i][j][r] * scale + ba;
                if (resid) v += resid[(size_t)row * N + col];
                if (outF)  outF[(size_t)row * N + col] = v;
                if (outB) {
                    bf16_t bv = (bf16_t)v;
                    if (transB) outB[(size_t)col * M + row] = bv;
                    else        outB[(size_t)row * N + col] = bv;
                }
            }
        }
    }
}

// ---------------------------------------------------------------------------
// Row-wise masked softmax over SEQ_LEN, in place on f32 scores; also writes
// bf16 copy of the probabilities for the attn@V GEMM.
// One 256-thread block per row; 16 columns per thread.
// ---------------------------------------------------------------------------
__global__ void softmax_mask_kernel(float* __restrict__ attn,
                                    const unsigned char* __restrict__ mask,
                                    bf16_t* __restrict__ attnB)
{
    const int row = blockIdx.x;
    const int t   = threadIdx.x;
    float* arow = attn + (size_t)row * SEQ_LEN;
    const unsigned char* mrow = mask + (size_t)row * SEQ_LEN;
    __shared__ float red[256];

    float vals[16];
    float mx = -INFINITY;
#pragma unroll
    for (int i = 0; i < 16; ++i) {
        const int c = t + i * 256;
        float v = mrow[c] ? -INFINITY : arow[c];
        vals[i] = v;
        mx = fmaxf(mx, v);
    }
    red[t] = mx; __syncthreads();
    for (int s = 128; s > 0; s >>= 1) {
        if (t < s) red[t] = fmaxf(red[t], red[t + s]);
        __syncthreads();
    }
    mx = red[0]; __syncthreads();

    float sum = 0.0f;
#pragma unroll
    for (int i = 0; i < 16; ++i) {
        float e = __expf(vals[i] - mx);
        vals[i] = e;
        sum += e;
    }
    red[t] = sum; __syncthreads();
    for (int s = 128; s > 0; s >>= 1) {
        if (t < s) red[t] += red[t + s];
        __syncthreads();
    }
    const float inv = 1.0f / red[0];
#pragma unroll
    for (int i = 0; i < 16; ++i) {
        const int c = t + i * 256;
        float p = vals[i] * inv;
        arow[c] = p;
        attnB[(size_t)row * SEQ_LEN + c] = (bf16_t)p;
    }
}

// ---------------------------------------------------------------------------
// Row-wise LayerNorm over DMODEL. One 256-thread block per row, 4 elems/thread.
// ---------------------------------------------------------------------------
__global__ void layernorm_kernel(const float* __restrict__ h,
                                 const float* __restrict__ gamma,
                                 const float* __restrict__ beta,
                                 float* __restrict__ out)
{
    const int row = blockIdx.x;
    const int t   = threadIdx.x;
    const float* hr = h + (size_t)row * DMODEL;
    __shared__ float red[256];

    float v[4];
    float s = 0.0f;
#pragma unroll
    for (int i = 0; i < 4; ++i) {
        v[i] = hr[t + i * 256];
        s += v[i];
    }
    red[t] = s; __syncthreads();
    for (int k = 128; k > 0; k >>= 1) {
        if (t < k) red[t] += red[t + k];
        __syncthreads();
    }
    const float mu = red[0] * (1.0f / DMODEL);
    __syncthreads();

    float sv = 0.0f;
#pragma unroll
    for (int i = 0; i < 4; ++i) {
        float d = v[i] - mu;
        sv += d * d;
    }
    red[t] = sv; __syncthreads();
    for (int k = 128; k > 0; k >>= 1) {
        if (t < k) red[t] += red[t + k];
        __syncthreads();
    }
    const float rstd = rsqrtf(red[0] * (1.0f / DMODEL) + 1e-5f);
#pragma unroll
    for (int i = 0; i < 4; ++i) {
        const int c = t + i * 256;
        out[(size_t)row * DMODEL + c] = (v[i] - mu) * rstd * gamma[c] + beta[c];
    }
}

// ---------------------------------------------------------------------------
// Orchestration
// ---------------------------------------------------------------------------
extern "C" void kernel_launch(void* const* d_in, const int* in_sizes, int n_in,
                              void* d_out, int out_size, void* d_ws, size_t ws_size,
                              hipStream_t stream)
{
    (void)in_sizes; (void)n_in; (void)out_size; (void)ws_size;

    const float* x    = (const float*)d_in[0];                 // [S, DM]
    const unsigned char* mask = (const unsigned char*)d_in[1]; // [S, S] bool
    const float* wq   = (const float*)d_in[2];
    const float* bq   = (const float*)d_in[3];
    const float* wk   = (const float*)d_in[4];
    const float* bk   = (const float*)d_in[5];
    const float* wv   = (const float*)d_in[6];
    const float* bv   = (const float*)d_in[7];
    const float* wo   = (const float*)d_in[8];
    const float* bo   = (const float*)d_in[9];
    const float* gamma = (const float*)d_in[10];
    const float* beta  = (const float*)d_in[11];

    float* outPtr = (float*)d_out;
    float* attnF  = outPtr + (size_t)SEQ_LEN * DMODEL;  // attention slab of d_out

    // Workspace layout (bf16 copies + f32 h)
    char* ws = (char*)d_ws;
    const size_t SZ_XD  = (size_t)SEQ_LEN * DMODEL * sizeof(bf16_t);  // 8 MiB
    const size_t SZ_W   = (size_t)DMODEL * DMODEL * sizeof(bf16_t);   // 2 MiB
    const size_t SZ_ATT = (size_t)SEQ_LEN * SEQ_LEN * sizeof(bf16_t); // 32 MiB
    bf16_t* xb    = (bf16_t*)(ws);                    ws += SZ_XD;
    bf16_t* wqT   = (bf16_t*)(ws);                    ws += SZ_W;
    bf16_t* wkT   = (bf16_t*)(ws);                    ws += SZ_W;
    bf16_t* wvT   = (bf16_t*)(ws);                    ws += SZ_W;
    bf16_t* woT   = (bf16_t*)(ws);                    ws += SZ_W;
    bf16_t* Qb    = (bf16_t*)(ws);                    ws += SZ_XD;
    bf16_t* Kb    = (bf16_t*)(ws);                    ws += SZ_XD;
    bf16_t* VTb   = (bf16_t*)(ws);                    ws += SZ_XD;   // [DM, S] transposed
    bf16_t* attnB = (bf16_t*)(ws);                    ws += SZ_ATT;
    bf16_t* ctxB  = (bf16_t*)(ws);                    ws += SZ_XD;
    float*  hF    = (float*)(ws);                     /* SEQ*DM f32 */

    const int THREADS = 256;   // 8 waves per block

    // 1) Convert activations and (transposed) weights to bf16
    cvt_bf16_kernel<<<(SEQ_LEN * DMODEL) / THREADS, THREADS, 0, stream>>>(x, xb, SEQ_LEN * DMODEL);
    transpose_cvt_kernel<<<(DMODEL * DMODEL) / THREADS, THREADS, 0, stream>>>(wq, wqT, DMODEL, DMODEL);
    transpose_cvt_kernel<<<(DMODEL * DMODEL) / THREADS, THREADS, 0, stream>>>(wk, wkT, DMODEL, DMODEL);
    transpose_cvt_kernel<<<(DMODEL * DMODEL) / THREADS, THREADS, 0, stream>>>(wv, wvT, DMODEL, DMODEL);
    transpose_cvt_kernel<<<(DMODEL * DMODEL) / THREADS, THREADS, 0, stream>>>(wo, woT, DMODEL, DMODEL);

    // Waves per GEMM: (M/64)*(N/32) 64x32 macro-tiles, 8 waves per block
    const int qkvBlocks   = ((SEQ_LEN / 64) * (DMODEL / 32) + 7) / 8;   // 256
    const int scoreBlocks = ((SEQ_LEN / 64) * (SEQ_LEN / 32) + 7) / 8;  // 1024

    // 2) QKV projections (bf16 WMMA, f32 accum). V stored transposed for attn@V.
    gemm_wmma_bf16_kernel<<<qkvBlocks, THREADS, 0, stream>>>(
        xb, DMODEL, wqT, DMODEL, bq, nullptr, nullptr, Qb, 0,
        SEQ_LEN, DMODEL, DMODEL, 1.0f);
    gemm_wmma_bf16_kernel<<<qkvBlocks, THREADS, 0, stream>>>(
        xb, DMODEL, wkT, DMODEL, bk, nullptr, nullptr, Kb, 0,
        SEQ_LEN, DMODEL, DMODEL, 1.0f);
    gemm_wmma_bf16_kernel<<<qkvBlocks, THREADS, 0, stream>>>(
        xb, DMODEL, wvT, DMODEL, bv, nullptr, nullptr, VTb, 1,
        SEQ_LEN, DMODEL, DMODEL, 1.0f);

    // 3) scores = Q @ K^T / 8  (K's row-major layout is already the "BT" layout)
    gemm_wmma_bf16_kernel<<<scoreBlocks, THREADS, 0, stream>>>(
        Qb, DMODEL, Kb, DMODEL, nullptr, nullptr, attnF, nullptr, 0,
        SEQ_LEN, SEQ_LEN, DMODEL, 0.125f);

    // 4) mask + softmax in place (f32 attention output) + bf16 copy
    softmax_mask_kernel<<<SEQ_LEN, THREADS, 0, stream>>>(attnF, mask, attnB);

    // 5) context = attn @ V + x (residual fused into epilogue), bf16 out
    gemm_wmma_bf16_kernel<<<qkvBlocks, THREADS, 0, stream>>>(
        attnB, SEQ_LEN, VTb, SEQ_LEN, nullptr, x, nullptr, ctxB, 0,
        SEQ_LEN, DMODEL, SEQ_LEN, 1.0f);

    // 6) h = context @ wo + bo (f32 out)
    gemm_wmma_bf16_kernel<<<qkvBlocks, THREADS, 0, stream>>>(
        ctxB, DMODEL, woT, DMODEL, bo, nullptr, hF, nullptr, 0,
        SEQ_LEN, DMODEL, DMODEL, 1.0f);

    // 7) LayerNorm -> output slab of d_out
    layernorm_kernel<<<SEQ_LEN, THREADS, 0, stream>>>(hF, gamma, beta, outPtr);
}
